// RelativeTransformerLinearClassifier_3092376453675
// MI455X (gfx1250) — matrix operand order repaired and checked by reading, not compiled, not run on hardware
//
#include <hip/hip_runtime.h>

// ---------------------------------------------------------------- constants
#define BB    8
#define TT    512
#define DD    1024
#define HH    16
#define HDIM  64
#define HFF   4096
#define LLY   4
#define NLL   128
#define BT    (BB * TT)   // 4096 rows
#define T2    (2 * TT)    // 1024 relative positions

typedef __attribute__((ext_vector_type(16))) __bf16 v16bf;
typedef __attribute__((ext_vector_type(8)))  float  v8f;

// =====================================================================
// Generic bf16-WMMA GEMM:  C(MxN) = A(MxK) @ B(KxN)  [+bias][leakyrelu]
// Software-pipelined: next K-tile's global loads issued before the WMMAs
// of the current tile, so load latency hides under the MMA work.
// B tile stored TRANSPOSED ([n][k]) so B-fragment reads are 2x ds_load_b128.
// epi: 0 = none, 1 = bias + leaky_relu(0.01), 2 = bias only
// Tile 128x128, K-step 32, 256 threads = 8 waves, each wave 32x64.
// =====================================================================
__global__ __launch_bounds__(256) void k_gemm(const float* __restrict__ A,
                                              const float* __restrict__ Bw,
                                              const float* __restrict__ bias,
                                              float* __restrict__ C,
                                              int M, int N, int K, int epi) {
  __shared__ __bf16 As[128][32 + 8];   // [m][k]
  __shared__ __bf16 Bs[128][32 + 8];   // [n][k]  (transposed!)

  const int tid  = threadIdx.x;
  const int lane = tid & 31;
  const int wid  = tid >> 5;
  const int wr   = wid & 3;    // wave row: rows wr*32 .. wr*32+31
  const int wc   = wid >> 2;   // wave col: cols wc*64 .. wc*64+63
  const int m16  = lane & 15;
  const int half = lane >> 4;
  const int bm   = blockIdx.y * 128;
  const int bn   = blockIdx.x * 128;

  v8f acc[2][4];
#pragma unroll
  for (int mf = 0; mf < 2; ++mf)
#pragma unroll
    for (int nf = 0; nf < 4; ++nf)
      acc[mf][nf] = (v8f){0.f, 0.f, 0.f, 0.f, 0.f, 0.f, 0.f, 0.f};

  // ---- prologue: load K-tile 0 into registers (8 batched b128 loads) ----
  float4 aR[4], bR[4];
#pragma unroll
  for (int r = 0; r < 4; ++r) {
    int idx = tid + r * 256;
    aR[r] = *(const float4*)(A + (size_t)(bm + (idx >> 3)) * K + ((idx & 7) << 2));
    bR[r] = *(const float4*)(Bw + (size_t)(idx >> 5) * N + bn + ((idx & 31) << 2));
  }

  for (int k0 = 0; k0 < K; k0 += 32) {
    // ---- cvt + store current tile regs -> LDS ----
#pragma unroll
    for (int r = 0; r < 4; ++r) {
      int idx  = tid + r * 256;
      int arow = idx >> 3, akc = (idx & 7) << 2;
      As[arow][akc + 0] = (__bf16)aR[r].x;  As[arow][akc + 1] = (__bf16)aR[r].y;
      As[arow][akc + 2] = (__bf16)aR[r].z;  As[arow][akc + 3] = (__bf16)aR[r].w;
      int brow = idx >> 5, bnc = (idx & 31) << 2;
      Bs[bnc + 0][brow] = (__bf16)bR[r].x;  Bs[bnc + 1][brow] = (__bf16)bR[r].y;
      Bs[bnc + 2][brow] = (__bf16)bR[r].z;  Bs[bnc + 3][brow] = (__bf16)bR[r].w;
    }
    __syncthreads();

    // ---- issue next tile's global loads now (hide under WMMAs) ----
    if (k0 + 32 < K) {
#pragma unroll
      for (int r = 0; r < 4; ++r) {
        int idx = tid + r * 256;
        aR[r] = *(const float4*)(A + (size_t)(bm + (idx >> 3)) * K + (k0 + 32) + ((idx & 7) << 2));
        bR[r] = *(const float4*)(Bw + (size_t)(k0 + 32 + (idx >> 5)) * N + bn + ((idx & 31) << 2));
      }
    }
    if (k0 + 64 < K) {
      __builtin_prefetch(A + (size_t)(bm + (tid >> 3)) * K + k0 + 64 + ((tid & 7) << 2), 0, 1);
      __builtin_prefetch(Bw + (size_t)(k0 + 64 + (tid >> 5)) * N + bn + ((tid & 31) << 2), 0, 1);
    }

    // ---- build fragments (16-bit A 16x32 / B 32x16; contiguous k = half*16 + i in both) ----
    v16bf afrag[2];
#pragma unroll
    for (int mf = 0; mf < 2; ++mf) {
      int m = wr * 32 + mf * 16 + m16;
      v16bf a;
#pragma unroll
      for (int v = 0; v < 8; ++v) {
        int s = v >> 2, g = v & 3;
        int kk = half * 16 + 2 * v;
        a[2 * v + 0] = As[m][kk];
        a[2 * v + 1] = As[m][kk + 1];
      }
      afrag[mf] = a;
    }
    v16bf bfrag[4];
#pragma unroll
    for (int nf = 0; nf < 4; ++nf) {
      int n = wc * 64 + nf * 16 + m16;
      const __bf16* bp = &Bs[n][half * 16];   // 16 contiguous bf16
      v16bf b;
#pragma unroll
      for (int v = 0; v < 16; ++v) b[v] = bp[v];
      bfrag[nf] = b;
    }
#pragma unroll
    for (int mf = 0; mf < 2; ++mf)
#pragma unroll
      for (int nf = 0; nf < 4; ++nf)
        acc[mf][nf] = __builtin_amdgcn_wmma_f32_16x16x32_bf16(
            false, afrag[mf], false, bfrag[nf], (short)0, acc[mf][nf], false, false);
    __syncthreads();
  }

  // ---- epilogue (C layout: M = v + 8*half, N = lane&15) ----
#pragma unroll
  for (int mf = 0; mf < 2; ++mf)
#pragma unroll
    for (int nf = 0; nf < 4; ++nf)
#pragma unroll
      for (int v = 0; v < 8; ++v) {
        int gr = bm + wr * 32 + mf * 16 + v + 8 * half;
        int gc = bn + wc * 64 + nf * 16 + m16;
        float val = acc[mf][nf][v];
        if (epi == 1) { val += bias[gc]; val = (val > 0.f) ? val : 0.01f * val; }
        else if (epi == 2) { val += bias[gc]; }
        C[(size_t)gr * N + gc] = val;
      }
}

// =====================================================================
// Relative sinusoidal position table: pos[l][d], l in [0,2T), d in [0,64)
// =====================================================================
__global__ void k_pos(float* __restrict__ pos) {
  int idx = blockIdx.x * blockDim.x + threadIdx.x;   // 2T * 32
  if (idx >= T2 * 32) return;
  int l = idx >> 5, i = idx & 31;
  float freq = __expf((float)i * (-logf(10000.f) / 31.f));
  float p = (float)(l - TT);
  pos[l * HDIM + i]      = __sinf(p * freq);
  pos[l * HDIM + 32 + i] = __cosf(p * freq);
}

// rwpos[l][h][pos_l] = r_w[l,h,:] . pos[pos_l,:]
__global__ void k_rwpos(const float* __restrict__ r_w,
                        const float* __restrict__ pos,
                        float* __restrict__ rwpos) {
  int idx = blockIdx.x * blockDim.x + threadIdx.x;   // LLY*HH*T2
  if (idx >= LLY * HH * T2) return;
  int l2  = idx & (T2 - 1);
  int hh  = (idx / T2) & (HH - 1);
  int lyr = idx / (T2 * HH);
  const float* rw = r_w + ((size_t)lyr * HH + hh) * HDIM;
  const float* pr = pos + (size_t)l2 * HDIM;
  float s = 0.f;
#pragma unroll
  for (int d = 0; d < HDIM; ++d) s += rw[d] * pr[d];
  rwpos[idx] = s;
}

// =====================================================================
// Residual + LayerNorm over D=1024: out = LN(base + delta) * g + b
// one block (256 threads) per row; in-place safe (block-local row).
// =====================================================================
__global__ __launch_bounds__(256) void k_ln(const float* __restrict__ base,
                                            const float* __restrict__ delta,
                                            const float* __restrict__ g,
                                            const float* __restrict__ bb,
                                            float* __restrict__ out) {
  __shared__ float red1[8], red2[8];
  const int row = blockIdx.x;
  const int tid = threadIdx.x;
  const int lane = tid & 31, wid = tid >> 5;
  float x[4], s = 0.f;
#pragma unroll
  for (int i = 0; i < 4; ++i) {
    int c = tid + i * 256;
    x[i] = base[(size_t)row * DD + c] + delta[(size_t)row * DD + c];
    s += x[i];
  }
#pragma unroll
  for (int off = 16; off; off >>= 1) s += __shfl_xor(s, off, 32);
  if (lane == 0) red1[wid] = s;
  __syncthreads();
  float mean = 0.f;
#pragma unroll
  for (int w = 0; w < 8; ++w) mean += red1[w];
  mean *= (1.f / (float)DD);
  float vs = 0.f;
#pragma unroll
  for (int i = 0; i < 4; ++i) { float d = x[i] - mean; vs += d * d; }
#pragma unroll
  for (int off = 16; off; off >>= 1) vs += __shfl_xor(vs, off, 32);
  if (lane == 0) red2[wid] = vs;
  __syncthreads();
  float var = 0.f;
#pragma unroll
  for (int w = 0; w < 8; ++w) var += red2[w];
  var *= (1.f / (float)DD);
  float inv = rsqrtf(var + 1e-5f);
#pragma unroll
  for (int i = 0; i < 4; ++i) {
    int c = tid + i * 256;
    out[(size_t)row * DD + c] = (x[i] - mean) * inv * g[c] + bb[c];
  }
}

// =====================================================================
// TENER relative attention, flash-style online softmax.
// 1 wave per (b, h, 16 query rows).  k-tiles of 64.
//   AC = (q + r_r) . k^T           (k = raw hidden per head)
//   BD[q,k] = (q . pos[T+k-q]) + rwpos[h, T+k-q]
// V tile staged in LDS transposed ([d][kc]) so P@V B-frags are ds_load_b128.
// =====================================================================
__global__ __launch_bounds__(32) void k_attn(const float* __restrict__ hbuf,
                                             const float* __restrict__ qv,
                                             const float* __restrict__ pos,
                                             const float* __restrict__ rwpos_l,
                                             const float* __restrict__ rr_l,
                                             const unsigned char* __restrict__ mask,
                                             float* __restrict__ out) {
  __shared__ float  s1_lds[16][80];        // BDf window tile
  __shared__ __bf16 p_lds[16][64 + 8];     // probabilities C->A relayout
  __shared__ __bf16 vs_lds[64][64 + 8];    // V tile transposed: [d][kc]

  const int bid  = blockIdx.x;
  const int qt   = bid & 31;               // T/16 = 32 query tiles
  const int hh   = (bid >> 5) & (HH - 1);
  const int b    = bid >> 9;
  const int lane = threadIdx.x;
  const int m16  = lane & 15;
  const int half = lane >> 4;
  const int q0   = qt * 16;

  // ---- loop-invariant A fragments: raw q and q + r_r (2 k-steps over d) ----
  v16bf aq[2], aqr[2];
  {
    const float* qrow = qv + ((size_t)(b * TT + q0 + m16)) * (2 * DD) + hh * HDIM;
    const float* rr   = rr_l + hh * HDIM;
#pragma unroll
    for (int s = 0; s < 2; ++s) {
      v16bf va, vb;
#pragma unroll
      for (int v = 0; v < 8; ++v) {
        int ss = v >> 2, gg = v & 3;
        int kk = s * 32 + half * 16 + 2 * v;
        float q0v = qrow[kk], q1v = qrow[kk + 1];
        va[2 * v + 0] = (__bf16)q0v;            va[2 * v + 1] = (__bf16)q1v;
        vb[2 * v + 0] = (__bf16)(q0v + rr[kk]); vb[2 * v + 1] = (__bf16)(q1v + rr[kk + 1]);
      }
      aq[s] = va; aqr[s] = vb;
    }
  }

  float mrow[8], lrow[8];
  v8f o[4];
#pragma unroll
  for (int v = 0; v < 8; ++v) { mrow[v] = -1e30f; lrow[v] = 0.f; }
#pragma unroll
  for (int nf = 0; nf < 4; ++nf)
    o[nf] = (v8f){0.f, 0.f, 0.f, 0.f, 0.f, 0.f, 0.f, 0.f};

  for (int kt = 0; kt < TT / 64; ++kt) {
    const int k0 = kt * 64;

    // ---- stage V tile (64 kc-rows x 64 d) into LDS transposed [d][kc] ----
    // batch the 16 b128 loads per row before the LDS scatter (no per-load waits)
#pragma unroll
    for (int rr2 = 0; rr2 < 2; ++rr2) {
      int kc = rr2 * 32 + lane;
      const float* vrow = qv + ((size_t)(b * TT + k0 + kc)) * (2 * DD) + DD + hh * HDIM;
      float4 vtmp[16];
#pragma unroll
      for (int dq = 0; dq < 16; ++dq) vtmp[dq] = *(const float4*)(vrow + dq * 4);
#pragma unroll
      for (int dq = 0; dq < 16; ++dq) {
        vs_lds[dq * 4 + 0][kc] = (__bf16)vtmp[dq].x;
        vs_lds[dq * 4 + 1][kc] = (__bf16)vtmp[dq].y;
        vs_lds[dq * 4 + 2][kc] = (__bf16)vtmp[dq].z;
        vs_lds[dq * 4 + 3][kc] = (__bf16)vtmp[dq].w;
      }
    }

    // ---- AC tile: S = (q+r_r) @ K^T  (K rows = raw hidden per head) ----
    v8f s_acc[4];
#pragma unroll
    for (int nf = 0; nf < 4; ++nf) {
      int kcol = k0 + nf * 16 + m16;
      v8f c = (v8f){0.f, 0.f, 0.f, 0.f, 0.f, 0.f, 0.f, 0.f};
#pragma unroll
      for (int s = 0; s < 2; ++s) {
        v16bf bf;
#pragma unroll
        for (int v = 0; v < 8; ++v) {
          int d = s * 32 + half * 16 + 2 * v;
          const float* kp = hbuf + ((size_t)(b * TT + kcol)) * DD + hh * HDIM + d;
          bf[2 * v + 0] = (__bf16)kp[0];
          bf[2 * v + 1] = (__bf16)kp[1];
        }
        c = __builtin_amdgcn_wmma_f32_16x16x32_bf16(false, aqr[s], false, bf, (short)0, c, false, false);
      }
      s_acc[nf] = c;
    }

    // ---- BDf window tile: S1(16x80) = q @ posWin^T,  l = l_base + col ----
    const int l_base = TT + k0 - q0 - 15;
#pragma unroll
    for (int nf = 0; nf < 5; ++nf) {
      int l = l_base + nf * 16 + m16;
      int lc = l < 0 ? 0 : (l > T2 - 1 ? T2 - 1 : l);
      v8f c = (v8f){0.f, 0.f, 0.f, 0.f, 0.f, 0.f, 0.f, 0.f};
#pragma unroll
      for (int s = 0; s < 2; ++s) {
        v16bf bf;
#pragma unroll
        for (int v = 0; v < 8; ++v) {
          int d = s * 32 + half * 16 + 2 * v;
          const float* pp = pos + (size_t)lc * HDIM + d;
          bf[2 * v + 0] = (__bf16)pp[0];
          bf[2 * v + 1] = (__bf16)pp[1];
        }
        c = __builtin_amdgcn_wmma_f32_16x16x32_bf16(false, aq[s], false, bf, (short)0, c, false, false);
      }
#pragma unroll
      for (int v = 0; v < 8; ++v)
        s1_lds[v + 8 * half][nf * 16 + m16] = c[v];
    }
    __syncthreads();

    // ---- gather diagonal shift, add rwpos, scale, mask ----
#pragma unroll
    for (int nf = 0; nf < 4; ++nf)
#pragma unroll
      for (int v = 0; v < 8; ++v) {
        int kj = nf * 16 + m16;
        int qi = v + 8 * half;
        int cdx = kj - qi + 15;                 // 0..78
        float sc = (s_acc[nf][v] + s1_lds[qi][cdx] + rwpos_l[hh * T2 + l_base + cdx]) * 0.125f;
        if (!mask[b * TT + k0 + kj]) sc = -1e9f;
        s_acc[nf][v] = sc;
      }

    // ---- online softmax update (row reduce across 16 lanes of each half) ----
    float mt[8];
#pragma unroll
    for (int v = 0; v < 8; ++v) {
      float m = s_acc[0][v];
#pragma unroll
      for (int nf = 1; nf < 4; ++nf) m = fmaxf(m, s_acc[nf][v]);
#pragma unroll
      for (int off = 1; off < 16; off <<= 1) m = fmaxf(m, __shfl_xor(m, off, 16));
      mt[v] = m;
    }
#pragma unroll
    for (int v = 0; v < 8; ++v) {
      float mnew = fmaxf(mrow[v], mt[v]);
      float corr = __expf(mrow[v] - mnew);
      mrow[v] = mnew;
      lrow[v] *= corr;
#pragma unroll
      for (int nf = 0; nf < 4; ++nf) o[nf][v] *= corr;
    }
#pragma unroll
    for (int v = 0; v < 8; ++v) {
      float ls = 0.f;
#pragma unroll
      for (int nf = 0; nf < 4; ++nf) {
        float p = __expf(s_acc[nf][v] - mrow[v]);
        s_acc[nf][v] = p;
        ls += p;
      }
#pragma unroll
      for (int off = 1; off < 16; off <<= 1) ls += __shfl_xor(ls, off, 16);
      lrow[v] += ls;
    }

    // ---- P: C-layout -> A-layout via LDS ----
#pragma unroll
    for (int nf = 0; nf < 4; ++nf)
#pragma unroll
      for (int v = 0; v < 8; ++v)
        p_lds[v + 8 * half][nf * 16 + m16] = (__bf16)s_acc[nf][v];
    __syncthreads();

    v16bf pa[2];
#pragma unroll
    for (int s = 0; s < 2; ++s) {
      v16bf a;
#pragma unroll
      for (int v = 0; v < 8; ++v) {
        int ss = v >> 2, gg = v & 3;
        int kk = s * 32 + half * 16 + 2 * v;
        a[2 * v + 0] = p_lds[m16][kk];
        a[2 * v + 1] = p_lds[m16][kk + 1];
      }
      pa[s] = a;
    }

    // ---- O += P @ V  (B-frags: 16 contiguous bf16 from vs_lds[d][.]) ----
#pragma unroll
    for (int nf = 0; nf < 4; ++nf) {
      int dcol = nf * 16 + m16;
      v8f c = o[nf];
#pragma unroll
      for (int s = 0; s < 2; ++s) {
        const __bf16* bp = &vs_lds[dcol][s * 32 + half * 16];
        v16bf bf;
#pragma unroll
        for (int v = 0; v < 16; ++v) bf[v] = bp[v];
        c = __builtin_amdgcn_wmma_f32_16x16x32_bf16(false, pa[s], false, bf, (short)0, c, false, false);
      }
      o[nf] = c;
    }
    __syncthreads();
  }

  // ---- finalize ----
#pragma unroll
  for (int nf = 0; nf < 4; ++nf)
#pragma unroll
    for (int v = 0; v < 8; ++v) {
      int qi = v + 8 * half;
      int d  = nf * 16 + m16;
      out[((size_t)(b * TT + q0 + qi)) * DD + hh * HDIM + d] = o[nf][v] / lrow[v];
    }
}

// =====================================================================
extern "C" void kernel_launch(void* const* d_in, const int* in_sizes, int n_in,
                              void* d_out, int out_size, void* d_ws, size_t ws_size,
                              hipStream_t stream) {
  const float* x    = (const float*)d_in[0];
  const unsigned char* mask = (const unsigned char*)d_in[1];
  const float* qv_w = (const float*)d_in[2];
  const float* r_r  = (const float*)d_in[3];
  const float* r_w  = (const float*)d_in[4];
  const float* ln1g = (const float*)d_in[5];
  const float* ln1b = (const float*)d_in[6];
  const float* w1   = (const float*)d_in[7];
  const float* b1   = (const float*)d_in[8];
  const float* w2   = (const float*)d_in[9];
  const float* b2   = (const float*)d_in[10];
  const float* ln2g = (const float*)d_in[11];
  const float* ln2b = (const float*)d_in[12];
  const float* clsw = (const float*)d_in[13];
  const float* clsb = (const float*)d_in[14];

  char* ws = (char*)d_ws;
  size_t o = 0;
  float* h     = (float*)(ws + o); o += (size_t)BT * DD * 4;       // 16 MB
  float* qv    = (float*)(ws + o); o += (size_t)BT * 2 * DD * 4;   // 32 MB
  float* tmp   = (float*)(ws + o); o += (size_t)BT * DD * 4;       // 16 MB
  float* ffmid = (float*)(ws + o); o += (size_t)BT * HFF * 4;      // 64 MB
  float* pos   = (float*)(ws + o); o += (size_t)T2 * HDIM * 4;
  float* rwpos = (float*)(ws + o); o += (size_t)LLY * HH * T2 * 4;

  hipMemcpyAsync(h, x, (size_t)BT * DD * 4, hipMemcpyDeviceToDevice, stream);
  k_pos<<<(T2 * 32 + 255) / 256, 256, 0, stream>>>(pos);
  k_rwpos<<<(LLY * HH * T2 + 255) / 256, 256, 0, stream>>>(r_w, pos, rwpos);

  for (int l = 0; l < LLY; ++l) {
    // qv = h @ qv_w[l]          (4096 x 2048 x 1024)
    k_gemm<<<dim3(2 * DD / 128, BT / 128), 256, 0, stream>>>(
        h, qv_w + (size_t)l * DD * 2 * DD, nullptr, qv, BT, 2 * DD, DD, 0);
    // attn_out = rel_attention
    k_attn<<<BB * HH * (TT / 16), 32, 0, stream>>>(
        h, qv, pos, rwpos + (size_t)l * HH * T2, r_r + (size_t)l * HH * HDIM, mask, tmp);
    // h = LN(h + attn_out)
    k_ln<<<BT, 256, 0, stream>>>(h, tmp, ln1g + (size_t)l * DD, ln1b + (size_t)l * DD, h);
    // ffmid = leaky_relu(h @ w1[l] + b1[l])   (4096 x 4096 x 1024)
    k_gemm<<<dim3(HFF / 128, BT / 128), 256, 0, stream>>>(
        h, w1 + (size_t)l * DD * HFF, b1 + (size_t)l * HFF, ffmid, BT, HFF, DD, 1);
    // ff_out = ffmid @ w2[l] + b2[l]          (4096 x 1024 x 4096)
    k_gemm<<<dim3(DD / 128, BT / 128), 256, 0, stream>>>(
        ffmid, w2 + (size_t)l * HFF * DD, b2 + (size_t)l * DD, tmp, BT, DD, HFF, 2);
    // h = LN(h + ff_out)
    k_ln<<<BT, 256, 0, stream>>>(h, tmp, ln2g + (size_t)l * DD, ln2b + (size_t)l * DD, h);
  }
  // logits = h @ cls_w + cls_b   (4096 x 128 x 1024)
  k_gemm<<<dim3(NLL / 128, BT / 128), 256, 0, stream>>>(
      h, clsw, clsb, (float*)d_out, BT, NLL, DD, 2);
}
